// TorchRoPEAttentionWithLoRA_16131897164479
// MI455X (gfx1250) — compile-verified, matrix operand order checked by reading, not run on hardware
//
#include <hip/hip_runtime.h>

// ---------------------------------------------------------------------------
// Types for CDNA5 WMMA (gfx1250, wave32)
// ---------------------------------------------------------------------------
typedef __bf16 bf16;
typedef __attribute__((ext_vector_type(8)))  __bf16 v8bf;
typedef __attribute__((ext_vector_type(16))) __bf16 v16bf;
typedef __attribute__((ext_vector_type(8)))  float  v8f;

#define HIDDEN   2048
#define SEQ      2048
#define BATCH    2
#define NQ       16
#define NKV      8
#define HD       128
#define MROWS    (BATCH * SEQ)            // 4096
#define NQKV     (NQ * HD + 2 * NKV * HD) // 4096  [q 2048 | k 1024 | v 1024]
#define LORA_SCALE 2.0f
#define QK_SCALE   0.08838834764831845f   // 1/sqrt(128)

__device__ __forceinline__ v16bf bfcat(v8bf lo, v8bf hi) {
    return __builtin_shufflevector(lo, hi, 0,1,2,3,4,5,6,7,8,9,10,11,12,13,14,15);
}

__device__ __forceinline__ v8f wmma_bf16(v16bf a, v16bf b, v8f c) {
    // (neg_a, A, neg_b, B, c_mod, C, reuse_a, reuse_b)
    return __builtin_amdgcn_wmma_f32_16x16x32_bf16(false, a, false, b, (short)0, c, false, false);
}

__device__ __forceinline__ v8f vzero8() {
    v8f z;
    #pragma unroll
    for (int i = 0; i < 8; ++i) z[i] = 0.0f;
    return z;
}

__device__ __forceinline__ float wavesum32(float v) {
    #pragma unroll
    for (int m = 1; m < 32; m <<= 1) v += __shfl_xor(v, m, 32);
    return v;
}

// A/B fragment K-element mapping (16-bit WMMA, §7.12.2):
// lane half h: elements 0..7  -> K = h*8 .. h*8+7   (contiguous, 16B)
//              elements 8..15 -> K = 16+h*8 .. +7   (contiguous, 16B)
__device__ __forceinline__ v16bf load_frag(const bf16* rowp /*points at K=half*8*/) {
    v8bf lo = *(const v8bf*)rowp;
    v8bf hi = *(const v8bf*)(rowp + 16);
    return bfcat(lo, hi);
}

// ---------------------------------------------------------------------------
// CDNA5 async copy: 32 bytes global -> LDS per lane (2 x B128, ASYNCcnt).
// The low 32 bits of a flat shared pointer are the LDS byte offset
// (addrspacecast(3->flat) = {SHARED_BASE, offset32}).
// dsaddr/memaddr both get INST_OFFSET added (ISA 08_async_tensor §4.4).
// ---------------------------------------------------------------------------
__device__ __forceinline__ void async_copy32(bf16* lds_dst, const bf16* gsrc) {
    unsigned lo = (unsigned)(uintptr_t)lds_dst;
    asm volatile(
        "global_load_async_to_lds_b128 %0, %1, off\n\t"
        "global_load_async_to_lds_b128 %0, %1, off offset:16"
        :: "v"(lo), "v"(gsrc)
        : "memory");
}

__device__ __forceinline__ void wait_async(int n) {
    if (n == 0)
        asm volatile("s_wait_asynccnt 0" ::: "memory");
    else
        asm volatile("s_wait_asynccnt 4" ::: "memory");
}

// ---------------------------------------------------------------------------
// fp32 -> bf16 conversion
// ---------------------------------------------------------------------------
__global__ void __launch_bounds__(256) cvt_f32_bf16(const float* __restrict__ src,
                                                    bf16* __restrict__ dst, size_t n) {
    size_t i = (size_t)blockIdx.x * blockDim.x + threadIdx.x;
    size_t stride = (size_t)gridDim.x * blockDim.x;
    for (; i < n; i += stride) dst[i] = (bf16)src[i];
}

// ---------------------------------------------------------------------------
// u = X(bf16, MxK) @ A^T(fp32, 8xK) -> U (Mx8 fp32); up to 3 A matrices
// one wave per row
// ---------------------------------------------------------------------------
__global__ void __launch_bounds__(256) lora_u3(
    const bf16* __restrict__ X, int K, int M,
    const float* __restrict__ A0, float* __restrict__ U0,
    const float* __restrict__ A1, float* __restrict__ U1,
    const float* __restrict__ A2, float* __restrict__ U2) {
    int t = threadIdx.x;
    int row = blockIdx.x * 8 + (t >> 5);
    int lane = t & 31;
    if (row >= M) return;
    const bf16* xr = X + (size_t)row * K;
    float a0[8], a1[8], a2[8];
    #pragma unroll
    for (int r = 0; r < 8; ++r) { a0[r] = 0.f; a1[r] = 0.f; a2[r] = 0.f; }
    for (int k = lane; k < K; k += 32) {
        float xv = (float)xr[k];
        #pragma unroll
        for (int r = 0; r < 8; ++r) a0[r] = fmaf(xv, A0[(size_t)r * K + k], a0[r]);
        if (A1) {
            #pragma unroll
            for (int r = 0; r < 8; ++r) a1[r] = fmaf(xv, A1[(size_t)r * K + k], a1[r]);
        }
        if (A2) {
            #pragma unroll
            for (int r = 0; r < 8; ++r) a2[r] = fmaf(xv, A2[(size_t)r * K + k], a2[r]);
        }
    }
    #pragma unroll
    for (int r = 0; r < 8; ++r) {
        a0[r] = wavesum32(a0[r]);
        if (A1) a1[r] = wavesum32(a1[r]);
        if (A2) a2[r] = wavesum32(a2[r]);
    }
    if (lane == 0) {
        #pragma unroll
        for (int r = 0; r < 8; ++r) {
            U0[(size_t)row * 8 + r] = a0[r];
            if (U1) U1[(size_t)row * 8 + r] = a1[r];
            if (U2) U2[(size_t)row * 8 + r] = a2[r];
        }
    }
}

// ---------------------------------------------------------------------------
// Shared GEMM core: C(128x128 f32) = A(128xK bf16) @ W^T, W is (N,K) bf16.
// 256 threads = 8 waves arranged 2(M) x 4(N); each wave: 4x2 tiles of 16x16.
// Double-buffered K-tiles staged with global_load_async_to_lds_b128.
// sA/sW each hold 2 buffers of 128x32 bf16 (8KB).
// ---------------------------------------------------------------------------
#define GEMM_BUF (128 * 32)

__device__ __forceinline__ void gemm_block(const bf16* __restrict__ Abase,
                                           const bf16* __restrict__ Wbase,
                                           int K, int m0, int n0,
                                           bf16* sA, bf16* sW, v8f acc[8]) {
    const int t = threadIdx.x;
    const int wave = t >> 5, lane = t & 31;
    const int half = lane >> 4, ln = lane & 15;
    const int wm = wave >> 2, wn = wave & 3;
    #pragma unroll
    for (int i = 0; i < 8; ++i) acc[i] = vzero8();

    const int r  = t >> 1;        // 0..127
    const int ch = (t & 1) << 4;  // 0 or 16 elements
    const bf16* ga = Abase + (size_t)(m0 + r) * K + ch;
    const bf16* gw = Wbase + (size_t)(n0 + r) * K + ch;
    bf16* la = sA + r * 32 + ch;
    bf16* lw = sW + r * 32 + ch;

    // prologue: tile 0 -> buffer 0 (4 async B128 ops per lane total)
    async_copy32(la, ga);
    async_copy32(lw, gw);

    int buf = 0;
    for (int k0 = 0; k0 < K; k0 += 32, buf ^= 1) {
        const int nxt = k0 + 32;
        if (nxt < K) {
            // prefetch next tile into the other buffer (safe: trailing barrier
            // of the previous iteration means no wave still reads it)
            async_copy32(la + (buf ^ 1) * GEMM_BUF, ga + nxt);
            async_copy32(lw + (buf ^ 1) * GEMM_BUF, gw + nxt);
            wait_async(4);   // the 4 older ops (current tile) are complete
        } else {
            wait_async(0);
        }
        __syncthreads();

        const bf16* cA = sA + buf * GEMM_BUF;
        const bf16* cW = sW + buf * GEMM_BUF;
        v16bf af[4], bfr[2];
        #pragma unroll
        for (int ms = 0; ms < 4; ++ms)
            af[ms] = load_frag(cA + (wm * 64 + ms * 16 + ln) * 32 + half * 8);
        #pragma unroll
        for (int ns = 0; ns < 2; ++ns)
            bfr[ns] = load_frag(cW + (wn * 32 + ns * 16 + ln) * 32 + half * 8);
        #pragma unroll
        for (int ms = 0; ms < 4; ++ms)
            #pragma unroll
            for (int ns = 0; ns < 2; ++ns)
                acc[ms * 2 + ns] = wmma_bf16(af[ms], bfr[ns], acc[ms * 2 + ns]);
        __syncthreads();
    }
}

// ---------------------------------------------------------------------------
// QKV projection + LoRA + RoPE + scatter into attention layouts
//   Wqkv: (4096, 2048) bf16 rows = [Wq | Wk | Wv]
//   q -> qb [b][h][s][d], k -> kb [b][h][s][d], v -> vT [b][h][d][s]
// ---------------------------------------------------------------------------
__global__ void __launch_bounds__(256) gemm_qkv_kernel(
    const bf16* __restrict__ xb, const bf16* __restrict__ Wqkv,
    const float* __restrict__ uq, const float* __restrict__ uk, const float* __restrict__ uv,
    const float* __restrict__ qB, const float* __restrict__ kB, const float* __restrict__ vB,
    bf16* __restrict__ qb, bf16* __restrict__ kb, bf16* __restrict__ vT) {
    __shared__ __align__(16) bf16 sA[2 * GEMM_BUF];
    __shared__ __align__(16) bf16 sW[2 * GEMM_BUF];
    const int m0 = blockIdx.x * 128;
    const int n0 = blockIdx.y * 128;
    v8f acc[8];
    gemm_block(xb, Wqkv, HIDDEN, m0, n0, sA, sW, acc);

    const int t = threadIdx.x;
    const int wave = t >> 5, lane = t & 31;
    const int half = lane >> 4, ln = lane & 15;
    const int wm = wave >> 2, wn = wave & 3;

    const bool isQ = (n0 < 2048);
    const bool isK = (n0 >= 2048) && (n0 < 3072);
    const float* U  = isQ ? uq : (isK ? uk : uv);
    const float* Bl = isQ ? qB : (isK ? kB : vB);
    const int regbase = isQ ? 0 : (isK ? 2048 : 3072);

    #pragma unroll
    for (int ms = 0; ms < 4; ++ms) {
        #pragma unroll
        for (int ns = 0; ns < 2; ++ns) {
            v8f a = acc[ms * 2 + ns];
            const int n   = n0 + wn * 32 + ns * 16 + ln;
            const int col = n - regbase;
            const int d   = col & 127;
            const int hd  = col >> 7;
            float bl[8];
            const float* blp = Bl + (size_t)col * 8;
            #pragma unroll
            for (int r = 0; r < 8; ++r) bl[r] = blp[r];

            // RoPE angle basis for this column (pair index d>>1)
            const float fr  = (float)(d & ~1) * (1.0f / 128.0f);
            const float invf = __expf(-9.210340371976184f * fr); // 10000^-fr
            #pragma unroll
            for (int v = 0; v < 8; ++v) {
                const int m = m0 + wm * 64 + ms * 16 + v + half * 8;
                const int b = m >> 11;
                const int s = m & 2047;
                float val = a[v];
                const float* up = U + (size_t)m * 8;
                float corr = 0.f;
                #pragma unroll
                for (int r = 0; r < 8; ++r) corr = fmaf(up[r], bl[r], corr);
                val += LORA_SCALE * corr;
                if (!isQ && !isK) {
                    // V: transposed store [b][h][d][s]
                    vT[((size_t)(b * NKV + hd) * HD + d) * SEQ + s] = (bf16)val;
                } else {
                    // RoPE on q/k; pair partner sits in lane^1 of same VGPR
                    float partner = __shfl_xor(val, 1, 32);
                    float ang = (float)s * invf;
                    float c, sn;
                    __sincosf(ang, &sn, &c);
                    val = (d & 1) ? fmaf(partner, sn, val * c)
                                  : fmaf(val, c, -partner * sn);
                    if (isQ) {
                        val *= QK_SCALE;
                        qb[((size_t)(b * NQ + hd) * SEQ + s) * HD + d] = (bf16)val;
                    } else {
                        kb[((size_t)(b * NKV + hd) * SEQ + s) * HD + d] = (bf16)val;
                    }
                }
            }
        }
    }
}

// ---------------------------------------------------------------------------
// Flash attention: one wave per (b, head, 16-row q tile).
// scores = q @ k^T (q pre-scaled), causal, online softmax, PV with WMMA.
// ---------------------------------------------------------------------------
__global__ void __launch_bounds__(256) attn_kernel(
    const bf16* __restrict__ qb, const bf16* __restrict__ kb,
    const bf16* __restrict__ vT, bf16* __restrict__ ctx) {
    __shared__ __align__(16) bf16 sP[8][16 * 32];
    const int t = threadIdx.x, wave = t >> 5, lane = t & 31;
    const int half = lane >> 4, ln = lane & 15;
    const int g  = blockIdx.x * 8 + wave;
    const int b  = g >> 11;
    const int h  = (g >> 7) & 15;
    const int qt = g & 127;
    const int q0 = qt * 16;
    const int kvh = h >> 1;
    const bf16* qp = qb + (size_t)(b * NQ + h) * SEQ * HD;
    const bf16* kp = kb + (size_t)(b * NKV + kvh) * SEQ * HD;
    const bf16* vp = vT + (size_t)(b * NKV + kvh) * HD * SEQ;
    bf16* sp = sP[wave];

    // q A-fragments over D=128 (4 x K32), resident in registers
    v16bf qa[4];
    #pragma unroll
    for (int kd = 0; kd < 4; ++kd)
        qa[kd] = load_frag(qp + (size_t)(q0 + ln) * HD + kd * 32 + half * 8);

    v8f o[8];
    #pragma unroll
    for (int dt = 0; dt < 8; ++dt) o[dt] = vzero8();
    float mx[8], sm[8];
    #pragma unroll
    for (int v = 0; v < 8; ++v) { mx[v] = -1e30f; sm[v] = 0.f; }

    for (int kc = 0; kc < q0 + 16; kc += 32) {
        v8f sc0 = vzero8(), sc1 = vzero8();
        #pragma unroll
        for (int kd = 0; kd < 4; ++kd) {
            v16bf k0f = load_frag(kp + (size_t)(kc + ln) * HD + kd * 32 + half * 8);
            sc0 = wmma_bf16(qa[kd], k0f, sc0);
            v16bf k1f = load_frag(kp + (size_t)(kc + 16 + ln) * HD + kd * 32 + half * 8);
            sc1 = wmma_bf16(qa[kd], k1f, sc1);
        }
        // causal mask + online softmax (rows live across 16 lanes of each half)
        float mnew[8];
        #pragma unroll
        for (int v = 0; v < 8; ++v) {
            const int qpos = q0 + v + half * 8;
            float s0 = ((kc + ln) <= qpos) ? sc0[v] : -1e30f;
            float s1 = ((kc + 16 + ln) <= qpos) ? sc1[v] : -1e30f;
            sc0[v] = s0; sc1[v] = s1;
            float mrow = fmaxf(s0, s1);
            #pragma unroll
            for (int msk = 1; msk < 16; msk <<= 1)
                mrow = fmaxf(mrow, __shfl_xor(mrow, msk, 32));
            mnew[v] = fmaxf(mx[v], mrow);
        }
        #pragma unroll
        for (int v = 0; v < 8; ++v) {
            float p0 = __expf(sc0[v] - mnew[v]);
            float p1 = __expf(sc1[v] - mnew[v]);
            sc0[v] = p0; sc1[v] = p1;
            float rs = p0 + p1;
            #pragma unroll
            for (int msk = 1; msk < 16; msk <<= 1) rs += __shfl_xor(rs, msk, 32);
            float alpha = __expf(mx[v] - mnew[v]);
            sm[v] = sm[v] * alpha + rs;
            mx[v] = mnew[v];
            #pragma unroll
            for (int dt = 0; dt < 8; ++dt) o[dt][v] *= alpha;
        }
        // transpose P (C-layout -> A-layout) through wave-private LDS
        #pragma unroll
        for (int v = 0; v < 8; ++v) {
            const int m = v + half * 8;
            sp[m * 32 + ln]      = (bf16)sc0[v];
            sp[m * 32 + 16 + ln] = (bf16)sc1[v];
        }
        asm volatile("s_wait_dscnt 0" ::: "memory");
        v16bf pa = load_frag(sp + ln * 32 + half * 8);
        #pragma unroll
        for (int dt = 0; dt < 8; ++dt) {
            v16bf vf = load_frag(vp + (size_t)(dt * 16 + ln) * SEQ + kc + half * 8);
            o[dt] = wmma_bf16(pa, vf, o[dt]);
        }
    }
    // normalize + store ctx [b][s][h*128 + d] (bf16)
    #pragma unroll
    for (int v = 0; v < 8; ++v) {
        const float inv = 1.0f / sm[v];
        const int qpos = q0 + v + half * 8;
        const size_t rowoff = ((size_t)(b * SEQ + qpos)) * (NQ * HD) + h * HD;
        #pragma unroll
        for (int dt = 0; dt < 8; ++dt)
            ctx[rowoff + dt * 16 + ln] = (bf16)(o[dt][v] * inv);
    }
}

// ---------------------------------------------------------------------------
// Output projection + LoRA -> fp32 d_out
// ---------------------------------------------------------------------------
__global__ void __launch_bounds__(256) gemm_out_kernel(
    const bf16* __restrict__ ctx, const bf16* __restrict__ Wob,
    const float* __restrict__ uo, const float* __restrict__ oB,
    float* __restrict__ out) {
    __shared__ __align__(16) bf16 sA[2 * GEMM_BUF];
    __shared__ __align__(16) bf16 sW[2 * GEMM_BUF];
    const int m0 = blockIdx.x * 128;
    const int n0 = blockIdx.y * 128;
    v8f acc[8];
    gemm_block(ctx, Wob, NQ * HD, m0, n0, sA, sW, acc);

    const int t = threadIdx.x;
    const int wave = t >> 5, lane = t & 31;
    const int half = lane >> 4, ln = lane & 15;
    const int wm = wave >> 2, wn = wave & 3;

    #pragma unroll
    for (int ms = 0; ms < 4; ++ms) {
        #pragma unroll
        for (int ns = 0; ns < 2; ++ns) {
            v8f a = acc[ms * 2 + ns];
            const int n = n0 + wn * 32 + ns * 16 + ln;
            float bl[8];
            const float* blp = oB + (size_t)n * 8;
            #pragma unroll
            for (int r = 0; r < 8; ++r) bl[r] = blp[r];
            #pragma unroll
            for (int v = 0; v < 8; ++v) {
                const int m = m0 + wm * 64 + ms * 16 + v + half * 8;
                const float* up = uo + (size_t)m * 8;
                float corr = 0.f;
                #pragma unroll
                for (int r = 0; r < 8; ++r) corr = fmaf(up[r], bl[r], corr);
                out[(size_t)m * HIDDEN + n] = a[v] + LORA_SCALE * corr;
            }
        }
    }
}

// ---------------------------------------------------------------------------
// Host launcher
// ---------------------------------------------------------------------------
extern "C" void kernel_launch(void* const* d_in, const int* in_sizes, int n_in,
                              void* d_out, int out_size, void* d_ws, size_t ws_size,
                              hipStream_t stream) {
    (void)in_sizes; (void)n_in; (void)out_size; (void)ws_size;
    const float* x  = (const float*)d_in[0];
    const float* Wq = (const float*)d_in[1];
    const float* Wk = (const float*)d_in[2];
    const float* Wv = (const float*)d_in[3];
    const float* Wo = (const float*)d_in[4];
    const float* qA = (const float*)d_in[5];
    const float* qB = (const float*)d_in[6];
    const float* kA = (const float*)d_in[7];
    const float* kB = (const float*)d_in[8];
    const float* vA = (const float*)d_in[9];
    const float* vB = (const float*)d_in[10];
    const float* oA = (const float*)d_in[11];
    const float* oB = (const float*)d_in[12];

    char* ws = (char*)d_ws;
    size_t off = 0;
    auto take = [&](size_t bytes) -> char* {
        char* p = ws + off;
        off += (bytes + 255) & ~(size_t)255;
        return p;
    };
    bf16* xb    = (bf16*)take((size_t)MROWS * HIDDEN * 2);
    bf16* Wqkvb = (bf16*)take((size_t)NQKV * HIDDEN * 2);
    bf16* Wob   = (bf16*)take((size_t)HIDDEN * (NQ * HD) * 2);
    float* uq   = (float*)take((size_t)MROWS * 8 * 4);
    float* uk   = (float*)take((size_t)MROWS * 8 * 4);
    float* uv   = (float*)take((size_t)MROWS * 8 * 4);
    float* uo   = (float*)take((size_t)MROWS * 8 * 4);
    bf16* qbuf  = (bf16*)take((size_t)BATCH * NQ * SEQ * HD * 2);
    bf16* kbuf  = (bf16*)take((size_t)BATCH * NKV * SEQ * HD * 2);
    bf16* vtbuf = (bf16*)take((size_t)BATCH * NKV * HD * SEQ * 2);
    bf16* ctx   = (bf16*)take((size_t)MROWS * (NQ * HD) * 2);

    // 1) convert inputs/weights to bf16 (Wq|Wk|Wv concatenated row-wise)
    cvt_f32_bf16<<<1024, 256, 0, stream>>>(x,  xb,    (size_t)MROWS * HIDDEN);
    cvt_f32_bf16<<<1024, 256, 0, stream>>>(Wq, Wqkvb, (size_t)2048 * HIDDEN);
    cvt_f32_bf16<<<512,  256, 0, stream>>>(Wk, Wqkvb + (size_t)2048 * HIDDEN, (size_t)1024 * HIDDEN);
    cvt_f32_bf16<<<512,  256, 0, stream>>>(Wv, Wqkvb + (size_t)3072 * HIDDEN, (size_t)1024 * HIDDEN);
    cvt_f32_bf16<<<1024, 256, 0, stream>>>(Wo, Wob,   (size_t)HIDDEN * (NQ * HD));

    // 2) rank-8 LoRA intermediates u = x @ A^T
    lora_u3<<<512, 256, 0, stream>>>(xb, HIDDEN, MROWS, qA, uq, kA, uk, vA, uv);

    // 3) fused QKV GEMM + LoRA + RoPE + scatter (grid: 32 M-tiles x 32 N-tiles)
    gemm_qkv_kernel<<<dim3(32, 32), 256, 0, stream>>>(
        xb, Wqkvb, uq, uk, uv, qB, kB, vB, qbuf, kbuf, vtbuf);

    // 4) flash attention (4096 waves)
    attn_kernel<<<512, 256, 0, stream>>>(qbuf, kbuf, vtbuf, ctx);

    // 5) u_o = ctx @ oA^T
    lora_u3<<<512, 256, 0, stream>>>(ctx, NQ * HD, MROWS, oA, uo,
                                     nullptr, nullptr, nullptr, nullptr);

    // 6) output projection + LoRA -> fp32
    gemm_out_kernel<<<dim3(32, 16), 256, 0, stream>>>(ctx, Wob, uo, oB, (float*)d_out);
}
